// AttentionBlock_Seq_31439160607267
// MI455X (gfx1250) — matrix operand order repaired and checked by reading, not compiled, not run on hardware
//
#include <hip/hip_runtime.h>

// MI455X / gfx1250 implementation.
// Roofline: ~220 GFLOP vs ~4 GB HBM traffic -> memory bound at 23.3 TB/s when
// matrix ops run on bf16 WMMA (16x16x32, f32 accum). fp32 WMMA (16x16x4) has
// 8x fewer FLOP/instr and would make the conv stages compute bound.
// Data movement: weight matrices (32KB each) are staged into LDS with
// global_load_async_to_lds_b128 (ASYNCcnt) so B-fragments come from ds_load
// and the vector-memory pipe is reserved for the random fp32 gathers/atomics
// that dominate HBM traffic. K3 double-buffers LDS so the async DMA of
// W_inv[k+1] overlaps the 32 WMMAs of step k. A-fragments for a whole row
// (4 K-steps) are loaded in one batch to maximize memory-level parallelism on
// the random gathers.

#define CN   128
#define NK   27
#define WSZ  (CN * CN)
#define EPSF 1e-5f

typedef __attribute__((ext_vector_type(16))) __bf16 v16bf;
typedef __attribute__((ext_vector_type(8)))  __bf16 v8bf;
typedef __attribute__((ext_vector_type(8)))  float  v8f;
typedef __attribute__((ext_vector_type(4)))  float  v4f;

static __device__ __forceinline__ v8f wmma_bf16(v16bf a, v16bf b, v8f c) {
  // D = A(16x32) * B(32x16) + C(16x16 f32)
  return __builtin_amdgcn_wmma_f32_16x16x32_bf16(false, a, false, b, (short)0, c,
                                                 false, false);
}

// Stage one 32KB (128x128 bf16) weight matrix into LDS with the async DMA
// path. 256 threads x 16B x 8 issues = 32KB. The 24-bit instruction offset is
// added to BOTH the LDS destination and the global source (ISA 15.18.3).
static __device__ __forceinline__ void stage_w32k(unsigned lds_base,
                                                  const __bf16* __restrict__ gsrc) {
  unsigned ldsa = lds_base + threadIdx.x * 16;
  unsigned goff = threadIdx.x * 16;
  asm volatile(
      "global_load_async_to_lds_b128 %0, %1, %2\n\t"
      "global_load_async_to_lds_b128 %0, %1, %2 offset:4096\n\t"
      "global_load_async_to_lds_b128 %0, %1, %2 offset:8192\n\t"
      "global_load_async_to_lds_b128 %0, %1, %2 offset:12288\n\t"
      "global_load_async_to_lds_b128 %0, %1, %2 offset:16384\n\t"
      "global_load_async_to_lds_b128 %0, %1, %2 offset:20480\n\t"
      "global_load_async_to_lds_b128 %0, %1, %2 offset:24576\n\t"
      "global_load_async_to_lds_b128 %0, %1, %2 offset:28672\n\t"
      :
      : "v"(ldsa), "v"(goff), "s"(gsrc)
      : "memory");
}

static __device__ __forceinline__ void wait_async_and_sync() {
  asm volatile("s_wait_asynccnt 0" ::: "memory");  // own wave's DMA done
  __syncthreads();                                 // everyone's DMA visible
}

// LDS byte offset of a shared variable (flat LDS aperture: addr[31:0] = offset)
#define LDS_OFF(p) ((unsigned)(uintptr_t)(p))

// All four A fragments (16x32 bf16 each, K = 0..127) of one fp32 row.
// Lane (m = lane&15) owns row m; half = lane>>4 selects K sub-blocks:
// frag ks, elems 0..7 -> K = 32ks + 8*half + e ; elems 8..15 -> +16.
// Loads are batched (16x global_load_b128) before any conversion so the
// memory pipe sees maximum outstanding loads (critical for random gathers).
static __device__ __forceinline__ void a_row_f32(const float* __restrict__ rowp,
                                                 int half, v16bf a[4]) {
  v4f t[16];
#pragma unroll
  for (int ks = 0; ks < 4; ++ks) {
    const float* p = rowp + ks * 32 + 8 * half;
    t[ks * 4 + 0] = *(const v4f*)p;
    t[ks * 4 + 1] = *(const v4f*)(p + 4);
    t[ks * 4 + 2] = *(const v4f*)(p + 16);
    t[ks * 4 + 3] = *(const v4f*)(p + 20);
  }
#pragma unroll
  for (int ks = 0; ks < 4; ++ks) {
#pragma unroll
    for (int e = 0; e < 4; ++e) {
      a[ks][e]      = (__bf16)t[ks * 4 + 0][e];
      a[ks][4 + e]  = (__bf16)t[ks * 4 + 1][e];
      a[ks][8 + e]  = (__bf16)t[ks * 4 + 2][e];
      a[ks][12 + e] = (__bf16)t[ks * 4 + 3][e];
    }
  }
}

// Same, from a bf16 row (8x global_load_b128, no conversion).
static __device__ __forceinline__ void a_row_bf16(const __bf16* __restrict__ rowp,
                                                  int half, v16bf a[4]) {
  v8bf t[8];
#pragma unroll
  for (int ks = 0; ks < 4; ++ks) {
    const __bf16* p = rowp + ks * 32 + 8 * half;
    t[ks * 2 + 0] = *(const v8bf*)p;
    t[ks * 2 + 1] = *(const v8bf*)(p + 16);
  }
#pragma unroll
  for (int ks = 0; ks < 4; ++ks) {
#pragma unroll
    for (int e = 0; e < 8; ++e) {
      a[ks][e]     = t[ks * 2 + 0][e];
      a[ks][8 + e] = t[ks * 2 + 1][e];
    }
  }
}

// B fragment (32x16 bf16) from an LDS copy of the weights, pre-transposed to
// Wt[n*CN + k]. Lane owns column n = n0*16 + (lane&15);
// elements 0..7 -> K = k0 + 8*(lane>>4) + e ; 8..15 -> +16  (two ds_load_b128 per fragment)
static __device__ __forceinline__ v16bf b_frag(const __bf16* __restrict__ Wt,
                                               int n0, int k0, int lane) {
  int n = n0 * 16 + (lane & 15);
  const __bf16* p = Wt + n * CN + k0 + 8 * (lane >> 4);
  v8bf lo = *(const v8bf*)p;
  v8bf hi = *(const v8bf*)(p + 16);
  v16bf b;
#pragma unroll
  for (int e = 0; e < 8; ++e) { b[e] = lo[e]; b[8 + e] = hi[e]; }
  return b;
}

// ---- K0: transpose + bf16-convert all weight matrices -----------------------
__global__ void k_prep(const float* __restrict__ Wg1, const float* __restrict__ Ws,
                       const float* __restrict__ Wcc, const float* __restrict__ Winv,
                       __bf16* __restrict__ Wg1t, __bf16* __restrict__ Wst,
                       __bf16* __restrict__ Wcct, __bf16* __restrict__ Winvt) {
  int idx = blockIdx.x * 256 + threadIdx.x;
  if (idx < WSZ) {
    int kk = idx >> 7, n = idx & 127;
    Wg1t[n * CN + kk] = (__bf16)Wg1[idx];
  } else if (idx < 2 * WSZ) {
    int j = idx - WSZ; int kk = j >> 7, n = j & 127;
    Wst[n * CN + kk] = (__bf16)Ws[j];
  } else if (idx < (2 + NK) * WSZ) {
    int j = idx - 2 * WSZ; int kt = j >> 14; int rem = j & (WSZ - 1);
    int ci = rem >> 7, co = rem & 127;
    Wcct[(size_t)kt * WSZ + co * CN + ci] = (__bf16)Wcc[j];
  } else if (idx < (2 + 2 * NK) * WSZ) {
    int j = idx - (2 + NK) * WSZ; int kt = j >> 14; int rem = j & (WSZ - 1);
    int ci = rem >> 7, co = rem & 127;
    Winvt[(size_t)kt * WSZ + co * CN + ci] = (__bf16)Winv[j];
  }
}

// ---- K1: phi = g_feats @ W_g1 -> bf16 ---------------------------------------
__global__ void k_gemm_phi(const float* __restrict__ A, const __bf16* __restrict__ Wt,
                           __bf16* __restrict__ out, int nrows) {
  __shared__ __align__(16) __bf16 wlds[WSZ];
  stage_w32k(LDS_OFF(wlds), Wt);
  int wave = threadIdx.x >> 5, lane = threadIdx.x & 31;
  int base = blockIdx.x * 128 + wave * 16;
  int m = lane & 15, half = lane >> 4;
  const float* rowp = A + (size_t)min(base + m, nrows - 1) * CN;
  v16bf af[4];
  a_row_f32(rowp, half, af);
  wait_async_and_sync();
  v8f acc[8] = {};
#pragma unroll
  for (int ks = 0; ks < 4; ++ks) {
#pragma unroll
    for (int n0 = 0; n0 < 8; ++n0)
      acc[n0] = wmma_bf16(af[ks], b_frag(wlds, n0, ks * 32, lane), acc[n0]);
  }
#pragma unroll
  for (int v = 0; v < 8; ++v) {
    int r = base + v + 8 * half;
    if (r < nrows) {
      __bf16* op = out + (size_t)r * CN + m;
#pragma unroll
      for (int n0 = 0; n0 < 8; ++n0) op[n0 * 16] = (__bf16)acc[n0][v];
    }
  }
}

// ---- K2: mid[rb_out[k][i]] += phi[i] @ W_cc[k]  (fp32 HW atomics) -----------
__global__ void k_conv_scatter(const __bf16* __restrict__ phi,
                               const __bf16* __restrict__ Wcct,
                               const int* __restrict__ rb_out,
                               float* __restrict__ mid, int N_, int M_) {
  __shared__ __align__(16) __bf16 wlds[WSZ];
  int k = blockIdx.y;                            // uniform per block
  stage_w32k(LDS_OFF(wlds), Wcct + (size_t)k * WSZ);
  int wave = threadIdx.x >> 5, lane = threadIdx.x & 31;
  int base = blockIdx.x * 128 + wave * 16;
  int m = lane & 15, half = lane >> 4;
  int i = base + m;
  int ro = (i < N_) ? rb_out[(size_t)k * N_ + i] : M_;  // M_ == invalid sentinel
  const __bf16* rowp = phi + (size_t)min(i, N_ - 1) * CN;
  v16bf af[4];
  a_row_bf16(rowp, half, af);
  wait_async_and_sync();
  v8f acc[8] = {};
#pragma unroll
  for (int ks = 0; ks < 4; ++ks) {
#pragma unroll
    for (int n0 = 0; n0 < 8; ++n0)
      acc[n0] = wmma_bf16(af[ks], b_frag(wlds, n0, ks * 32, lane), acc[n0]);
  }
  // Accumulator layout: acc[n0][v] = D(M = v + 8*half, N = n0*16 + m)
#pragma unroll
  for (int v = 0; v < 8; ++v) {
    int ro_v = __shfl(ro, v + 8 * half, 32);   // rb_out of tile row v+8*half
    if (ro_v < M_) {
      float* op = mid + (size_t)ro_v * CN + m;
#pragma unroll
      for (int n0 = 0; n0 < 8; ++n0) unsafeAtomicAdd(op + n0 * 16, acc[n0][v]);
    }
  }
}

// ---- K3: inv[i] = sum_k mid[rb_out[k][i]] @ W_inv[k]  (+ BN1 stats) ---------
// Double-buffered LDS weights: async DMA of W_inv[k+1] overlaps step k WMMAs.
__global__ void k_conv_gather(const float* __restrict__ mid,
                              const __bf16* __restrict__ Winvt,
                              const int* __restrict__ rb_out,
                              float* __restrict__ inv,
                              float* __restrict__ sums, int N_, int M_) {
  __shared__ __align__(16) __bf16 wlds[2][WSZ];
  __shared__ float red[2][8][CN];
  stage_w32k(LDS_OFF(&wlds[0][0]), Winvt);
  int wave = threadIdx.x >> 5, lane = threadIdx.x & 31;
  int base = blockIdx.x * 128 + wave * 16;
  int m = lane & 15, half = lane >> 4;
  int i = base + m;
  v8f acc[8] = {};
  int ro = (i < N_) ? rb_out[i] : M_;          // k = 0 ; row M_ is all-zero
  wait_async_and_sync();
#pragma unroll 1
  for (int k = 0; k < NK; ++k) {
    if (k + 1 < NK)   // target buffer was last read at k-1; barrier covered it
      stage_w32k(LDS_OFF(&wlds[(k + 1) & 1][0]), Winvt + (size_t)(k + 1) * WSZ);
    // Batched gather of the lane's half-row: 16 outstanding global_load_b128.
    v16bf af[4];
    a_row_f32(mid + (size_t)ro * CN, half, af);
    int ron = M_;
    if (k + 1 < NK) {
      ron = (i < N_) ? rb_out[(size_t)(k + 1) * N_ + i] : M_;
      __builtin_prefetch(mid + (size_t)ron * CN, 0, 0);   // global_prefetch_b8
    }
    const __bf16* Wt = &wlds[k & 1][0];
#pragma unroll
    for (int ks = 0; ks < 4; ++ks) {
#pragma unroll
      for (int n0 = 0; n0 < 8; ++n0)
        acc[n0] = wmma_bf16(af[ks], b_frag(Wt, n0, ks * 32, lane), acc[n0]);
    }
    ro = ron;
    wait_async_and_sync();   // next weights resident; all reads of k done
  }
  // Store inv + per-column sum / sum-of-squares for BN1
#pragma unroll
  for (int n0 = 0; n0 < 8; ++n0) {
    float s = 0.f, q = 0.f;
#pragma unroll
    for (int v = 0; v < 8; ++v) {
      int r = base + v + 8 * half;
      if (r < N_) {
        float t = acc[n0][v];
        inv[(size_t)r * CN + n0 * 16 + m] = t;
        s += t; q += t * t;
      }
    }
    s += __shfl_xor(s, 16, 32);
    q += __shfl_xor(q, 16, 32);
    if (lane < 16) { red[0][wave][n0 * 16 + lane] = s; red[1][wave][n0 * 16 + lane] = q; }
  }
  __syncthreads();
  if (threadIdx.x < CN) {
    float s = 0.f, q = 0.f;
#pragma unroll
    for (int w8 = 0; w8 < 8; ++w8) { s += red[0][w8][threadIdx.x]; q += red[1][w8][threadIdx.x]; }
    unsafeAtomicAdd(&sums[threadIdx.x], s);
    unsafeAtomicAdd(&sums[CN + threadIdx.x], q);
  }
}

// ---- K4: s = x_feats @ W_s -> fp32 (+ BN2 stats) ----------------------------
__global__ void k_gemm_theta(const float* __restrict__ A, const __bf16* __restrict__ Wt,
                             float* __restrict__ out, float* __restrict__ sums,
                             int nrows) {
  __shared__ __align__(16) __bf16 wlds[WSZ];
  __shared__ float red[2][8][CN];
  stage_w32k(LDS_OFF(wlds), Wt);
  int wave = threadIdx.x >> 5, lane = threadIdx.x & 31;
  int base = blockIdx.x * 128 + wave * 16;
  int m = lane & 15, half = lane >> 4;
  const float* rowp = A + (size_t)min(base + m, nrows - 1) * CN;
  v16bf af[4];
  a_row_f32(rowp, half, af);
  wait_async_and_sync();
  v8f acc[8] = {};
#pragma unroll
  for (int ks = 0; ks < 4; ++ks) {
#pragma unroll
    for (int n0 = 0; n0 < 8; ++n0)
      acc[n0] = wmma_bf16(af[ks], b_frag(wlds, n0, ks * 32, lane), acc[n0]);
  }
#pragma unroll
  for (int n0 = 0; n0 < 8; ++n0) {
    float s = 0.f, q = 0.f;
#pragma unroll
    for (int v = 0; v < 8; ++v) {
      int r = base + v + 8 * half;
      if (r < nrows) {
        float t = acc[n0][v];
        out[(size_t)r * CN + n0 * 16 + m] = t;
        s += t; q += t * t;
      }
    }
    s += __shfl_xor(s, 16, 32);
    q += __shfl_xor(q, 16, 32);
    if (lane < 16) { red[0][wave][n0 * 16 + lane] = s; red[1][wave][n0 * 16 + lane] = q; }
  }
  __syncthreads();
  if (threadIdx.x < CN) {
    float s = 0.f, q = 0.f;
#pragma unroll
    for (int w8 = 0; w8 < 8; ++w8) { s += red[0][w8][threadIdx.x]; q += red[1][w8][threadIdx.x]; }
    unsafeAtomicAdd(&sums[threadIdx.x], s);
    unsafeAtomicAdd(&sums[CN + threadIdx.x], q);
  }
}

// ---- K5: finalize BN1/BN2 statistics ----------------------------------------
__global__ void k_bn_finalize(const float* __restrict__ sums, float* __restrict__ stats,
                              int N_) {
  int c = threadIdx.x;
  float invN = 1.0f / (float)N_;
  float mu1 = sums[c] * invN;
  float v1  = fmaxf(sums[CN + c] * invN - mu1 * mu1, 0.f);
  float mu2 = sums[2 * CN + c] * invN;
  float v2  = fmaxf(sums[3 * CN + c] * invN - mu2 * mu2, 0.f);
  stats[c] = mu1;           stats[CN + c] = rsqrtf(v1 + EPSF);
  stats[2 * CN + c] = mu2;  stats[3 * CN + c] = rsqrtf(v2 + EPSF);
}

// ---- K6: z_i = relu(relu(bn1(inv)) + relu(bn2(s))) . w_up  (+ BN3 stats) ----
__global__ void k_fuse_z(const float* __restrict__ inv, const float* __restrict__ s,
                         const float* __restrict__ stats,
                         const float* __restrict__ g1, const float* __restrict__ b1,
                         const float* __restrict__ g2, const float* __restrict__ b2,
                         const float* __restrict__ wup,
                         float* __restrict__ z, float* __restrict__ sums, int N_) {
  __shared__ float red[8][2];
  int wave = threadIdx.x >> 5, lane = threadIdx.x & 31;
  float mu1[4], rs1[4], G1[4], B1[4], mu2[4], rs2[4], G2[4], B2[4], WU[4];
#pragma unroll
  for (int j = 0; j < 4; ++j) {
    int c = lane + 32 * j;
    mu1[j] = stats[c];          rs1[j] = stats[CN + c];
    mu2[j] = stats[2 * CN + c]; rs2[j] = stats[3 * CN + c];
    G1[j] = g1[c]; B1[j] = b1[c]; G2[j] = g2[c]; B2[j] = b2[c]; WU[j] = wup[c];
  }
  int gw = blockIdx.x * 8 + wave, nw = gridDim.x * 8;
  float zs = 0.f, zq = 0.f;
  for (int r = gw; r < N_; r += nw) {
    float d = 0.f;
#pragma unroll
    for (int j = 0; j < 4; ++j) {
      int c = lane + 32 * j;
      float a = fmaxf((inv[(size_t)r * CN + c] - mu1[j]) * rs1[j] * G1[j] + B1[j], 0.f);
      float t = fmaxf((s[(size_t)r * CN + c] - mu2[j]) * rs2[j] * G2[j] + B2[j], 0.f);
      d += fmaxf(a + t, 0.f) * WU[j];
    }
#pragma unroll
    for (int o = 16; o >= 1; o >>= 1) d += __shfl_xor(d, o, 32);
    if (lane == 0) { z[r] = d; zs += d; zq += d * d; }
  }
  if (lane == 0) { red[wave][0] = zs; red[wave][1] = zq; }
  __syncthreads();
  if (threadIdx.x == 0) {
    float a = 0.f, b = 0.f;
#pragma unroll
    for (int w8 = 0; w8 < 8; ++w8) { a += red[w8][0]; b += red[w8][1]; }
    unsafeAtomicAdd(&sums[4 * CN], a);
    unsafeAtomicAdd(&sums[4 * CN + 1], b);
  }
}

// ---- K7: out = (sigmoid(bn3(z)) * W_att + b_att) * x_feats ------------------
__global__ void k_output(const float* __restrict__ z, const float* __restrict__ sums,
                         const float* __restrict__ g3, const float* __restrict__ b3,
                         const float* __restrict__ Watt, const float* __restrict__ batt,
                         const float* __restrict__ x, float* __restrict__ out, int N_) {
  float invN = 1.0f / (float)N_;
  float mu3 = sums[4 * CN] * invN;
  float rs3 = rsqrtf(fmaxf(sums[4 * CN + 1] * invN - mu3 * mu3, 0.f) + EPSF);
  float G3 = g3[0], B3 = b3[0];
  size_t total = (size_t)N_ * CN;
  size_t stride = (size_t)gridDim.x * blockDim.x;
  for (size_t idx = (size_t)blockIdx.x * blockDim.x + threadIdx.x; idx < total;
       idx += stride) {
    int r = (int)(idx >> 7), c = (int)(idx & (CN - 1));
    float t = (z[r] - mu3) * rs3 * G3 + B3;
    float att = 1.0f / (1.0f + __expf(-t));
    out[idx] = (att * Watt[c] + batt[c]) * x[idx];
  }
}

extern "C" void kernel_launch(void* const* d_in, const int* in_sizes, int n_in,
                              void* d_out, int out_size, void* d_ws, size_t ws_size,
                              hipStream_t stream) {
  const float* g_feats = (const float*)d_in[0];
  const float* x_feats = (const float*)d_in[1];
  const int*   rb_out  = (const int*)d_in[3];   // rb_in unused: rb_in[k][i]==i when valid
  const float* Wg1  = (const float*)d_in[5];
  const float* Wcc  = (const float*)d_in[6];
  const float* Winv = (const float*)d_in[7];
  const float* bn1g = (const float*)d_in[8];
  const float* bn1b = (const float*)d_in[9];
  const float* Ws   = (const float*)d_in[10];
  const float* bn2g = (const float*)d_in[11];
  const float* bn2b = (const float*)d_in[12];
  const float* wup  = (const float*)d_in[13];
  const float* bn3g = (const float*)d_in[14];
  const float* bn3b = (const float*)d_in[15];
  const float* Watt = (const float*)d_in[16];
  const float* batt = (const float*)d_in[17];
  int N_ = in_sizes[0] / CN;
  int M_ = in_sizes[4];

  char* wsb = (char*)d_ws;
  size_t off = 0;
  auto carve = [&](size_t bytes) {
    char* p = wsb + off;
    off = (off + bytes + 255) & ~(size_t)255;
    return p;
  };
  __bf16* Wg1t  = (__bf16*)carve((size_t)WSZ * 2);
  __bf16* Wst   = (__bf16*)carve((size_t)WSZ * 2);
  __bf16* Wcct  = (__bf16*)carve((size_t)NK * WSZ * 2);
  __bf16* Winvt = (__bf16*)carve((size_t)NK * WSZ * 2);
  float*  sums  = (float*)carve((4 * CN + 2) * sizeof(float));
  float*  stats = (float*)carve(4 * CN * sizeof(float));
  __bf16* phi   = (__bf16*)carve((size_t)N_ * CN * 2);
  float*  inv   = (float*)carve((size_t)N_ * CN * 4);
  float*  sfeat = (float*)carve((size_t)N_ * CN * 4);
  float*  z     = (float*)carve((size_t)N_ * 4);
  float*  mid   = (float*)carve((size_t)(M_ + 1) * CN * 4);  // row M_ stays zero

  hipMemsetAsync(sums, 0, (4 * CN + 2) * sizeof(float), stream);
  hipMemsetAsync(mid, 0, (size_t)(M_ + 1) * CN * sizeof(float), stream);

  int prep_total = (2 + 2 * NK) * WSZ;
  k_prep<<<(prep_total + 255) / 256, 256, 0, stream>>>(Wg1, Ws, Wcc, Winv,
                                                       Wg1t, Wst, Wcct, Winvt);

  int tb = (N_ + 127) / 128;   // 256 threads = 8 waves, 16 rows per wave
  k_gemm_phi<<<tb, 256, 0, stream>>>(g_feats, Wg1t, phi, N_);
  dim3 gs(tb, NK);
  k_conv_scatter<<<gs, 256, 0, stream>>>(phi, Wcct, rb_out, mid, N_, M_);
  k_conv_gather<<<tb, 256, 0, stream>>>(mid, Winvt, rb_out, inv, sums, N_, M_);
  k_gemm_theta<<<tb, 256, 0, stream>>>(x_feats, Wst, sfeat, sums + 2 * CN, N_);
  k_bn_finalize<<<1, CN, 0, stream>>>(sums, stats, N_);
  k_fuse_z<<<tb, 256, 0, stream>>>(inv, sfeat, stats, bn1g, bn1b, bn2g, bn2b,
                                   wup, z, sums, N_);
  k_output<<<1024, 256, 0, stream>>>(z, sums, bn3g, bn3b, Watt, batt, x_feats,
                                     (float*)d_out, N_);
}
